// VectorQuantizer_43078521979117
// MI455X (gfx1250) — matrix operand order repaired and checked by reading, not compiled, not run on hardware
//
#include <hip/hip_runtime.h>

typedef __attribute__((ext_vector_type(16))) __bf16 v16bf;
typedef __attribute__((ext_vector_type(8)))  float  v8f;
typedef __attribute__((ext_vector_type(4)))  int    v4i;

#define NROWS 32768      // 32*32*32
#define KCODES 2048
#define DDIM 128
#define GATHER_BLOCKS (NROWS / 2)

// vq_argmin geometry: 512 threads = 16 waves, 16 rows/wave -> 256 rows/WG
#define AM_THREADS 512
#define AM_WAVES   16
#define ROWS_PER_WG (AM_WAVES * 16)
#define NPANEL 32                 // 2048 cols / 64 cols-per-panel
#define TILES_PER_PANEL 4
#define TILE_ELEMS 4096           // 8 chunks(var*4+s) * 32 lanes * 16 elems (bf16)
#define PANEL_ELEMS (TILES_PER_PANEL * TILE_ELEMS)   // 16384 bf16 = 32 KB

#if defined(__gfx1250__) && __has_builtin(__builtin_amdgcn_global_load_async_to_lds_b128) && __has_builtin(__builtin_amdgcn_s_wait_asynccnt)
#define USE_ASYNC_LDS 1
typedef __attribute__((address_space(1))) v4i* as1_v4i_p;
typedef __attribute__((address_space(3))) v4i* as3_v4i_p;
#else
#define USE_ASYNC_LDS 0
#endif

// ---------------------------------------------------------------------------
// Kernel 1: per-code half-norms, fp32 E^T, and hi/lo bf16 B operands
// pre-swizzled into this kernel's contiguous-k WMMA B lane order (lanes 0-15: K=0..15,
// lanes 16-31: K=16..31; elem e -> K = (lane>>4)*16 + e).
// Bsw element index: g*4096 + (var*4 + s)*512 + lane*16 + e
//   g = col>>4 (global 16-col tile), s = d>>5 (32-deep k-step),
//   var: 0 = hi, 1 = lo
// ---------------------------------------------------------------------------
__global__ __launch_bounds__(256) void vq_prep(const float* __restrict__ E,
                                               float* __restrict__ hn,
                                               float* __restrict__ ET,
                                               __bf16* __restrict__ Bsw) {
  int k = blockIdx.x * 256 + threadIdx.x;   // 0..2047
  const int g = k >> 4;
  float sum = 0.f;
  #pragma unroll 4
  for (int d = 0; d < DDIM; ++d) {
    float v = E[(size_t)d * KCODES + k];
    sum += v * v;
    ET[(size_t)k * DDIM + d] = v;
    __bf16 hi = (__bf16)v;
    __bf16 lo = (__bf16)(v - (float)hi);
    int s    = d >> 5;
    int lane = (k & 15) + (((d >> 4) & 1) << 4);
    int e    = d & 15;
    size_t base = (size_t)g * TILE_ELEMS + (size_t)lane * 16 + e;
    Bsw[base + (0 * 4 + s) * 512] = hi;
    Bsw[base + (1 * 4 + s) * 512] = lo;
  }
  hn[k] = 0.5f * sum;
}

// ---------------------------------------------------------------------------
// Panel stage: contiguous 32 KB global -> LDS copy (async if available)
// ---------------------------------------------------------------------------
__device__ __forceinline__ void stage_panel(const __bf16* __restrict__ Bsw,
                                            __bf16* dstLds, int panel, int t) {
  const uint4* src = (const uint4*)(Bsw + (size_t)panel * PANEL_ELEMS);
  uint4* dst = (uint4*)dstLds;
#if USE_ASYNC_LDS
  #pragma unroll
  for (int i = 0; i < 4; ++i)
    __builtin_amdgcn_global_load_async_to_lds_b128(
        (as1_v4i_p)(src + i * AM_THREADS + t),
        (as3_v4i_p)(dst + i * AM_THREADS + t),
        0, 0);
#else
  #pragma unroll
  for (int i = 0; i < 4; ++i)
    dst[i * AM_THREADS + t] = src[i * AM_THREADS + t];
#endif
}

__device__ __forceinline__ void wait_stage() {
#if USE_ASYNC_LDS
  __builtin_amdgcn_s_wait_asynccnt(0);
#endif
}

// ---------------------------------------------------------------------------
// Kernel 2: fused split-bf16 WMMA GEMM (x @ E) + per-row argmin of (hn - dot)
// dot = x_hi*e_hi + x_hi*e_lo + x_lo*e_hi  (fp32-grade accuracy, 3x WMMA)
// Grid: 128 blocks x 512 threads (16 waves). Wave: 16 rows, all 2048 cols.
// Double-buffered LDS B panels (2 x 32 KB), async copy overlapped w/ WMMA.
// ---------------------------------------------------------------------------
__global__ __launch_bounds__(AM_THREADS) void vq_argmin(
    const float* __restrict__ x, const __bf16* __restrict__ Bsw,
    const float* __restrict__ hn, int* __restrict__ enc) {
  __shared__ __align__(32) __bf16 Bp[2][PANEL_ELEMS];   // 64 KB

  const int t     = threadIdx.x;
  const int lane  = t & 31;
  const int wave  = t >> 5;
  const int lrow  = lane & 15;
  const int khalf = lane >> 4;
  const int row0  = blockIdx.x * ROWS_PER_WG + wave * 16;

  // ---- A tile (hi/lo) in registers, bf16 A layout:
  // lane m = l&15; elem e -> K = kh*16+e
  v16bf a_hi[4], a_lo[4];
  {
    const float* xr = x + (size_t)(row0 + lrow) * DDIM;
    #pragma unroll
    for (int s = 0; s < 4; ++s) {
      const int d0 = s * 32 + khalf * 16;
      const int d1 = s * 32 + khalf * 16 + 8;
      #pragma unroll
      for (int e = 0; e < 8; ++e) {
        float v0 = xr[d0 + e], v1 = xr[d1 + e];
        __bf16 h0 = (__bf16)v0, h1 = (__bf16)v1;
        a_hi[s][e]     = h0;  a_lo[s][e]     = (__bf16)(v0 - (float)h0);
        a_hi[s][e + 8] = h1;  a_lo[s][e + 8] = (__bf16)(v1 - (float)h1);
      }
    }
  }

  float bestV[8];
  int   bestI[8];
  #pragma unroll
  for (int r = 0; r < 8; ++r) { bestV[r] = 3.4e38f; bestI[r] = 0; }

  stage_panel(Bsw, &Bp[0][0], 0, t);
  wait_stage();
  __syncthreads();

  for (int p = 0; p < NPANEL; ++p) {
    const int buf = p & 1;
    if (p + 1 < NPANEL) stage_panel(Bsw, &Bp[buf ^ 1][0], p + 1, t);

    #pragma unroll
    for (int tau = 0; tau < TILES_PER_PANEL; ++tau) {
      const __bf16* tb = &Bp[buf][tau * TILE_ELEMS];
      v8f acc = {};
      #pragma unroll
      for (int s = 0; s < 4; ++s) {
        const v16bf bh = *(const v16bf*)(tb + (0 * 4 + s) * 512 + lane * 16);
        const v16bf bl = *(const v16bf*)(tb + (1 * 4 + s) * 512 + lane * 16);
        acc = __builtin_amdgcn_wmma_f32_16x16x32_bf16(false, a_hi[s], false, bh,
                                                      (short)0, acc, false, false);
        acc = __builtin_amdgcn_wmma_f32_16x16x32_bf16(false, a_hi[s], false, bl,
                                                      (short)0, acc, false, false);
        acc = __builtin_amdgcn_wmma_f32_16x16x32_bf16(false, a_lo[s], false, bh,
                                                      (short)0, acc, false, false);
      }
      const int n = (p * TILES_PER_PANEL + tau) * 16 + lrow;  // lane's column
      const float h = hn[n];
      #pragma unroll
      for (int r = 0; r < 8; ++r) {                 // rows m = khalf*8 + r
        float v = h - acc[r];
        bool lt = v < bestV[r];
        bestI[r] = lt ? n : bestI[r];
        bestV[r] = lt ? v : bestV[r];
      }
    }
    wait_stage();      // next panel's copy complete (this wave)
    __syncthreads();   // all waves done consuming buf before it is re-staged
  }

  // ---- cross-lane argmin within each 16-lane half (wave32 shuffles) ----
  #pragma unroll
  for (int m = 1; m <= 8; m <<= 1) {
    #pragma unroll
    for (int r = 0; r < 8; ++r) {
      float ov = __shfl_xor(bestV[r], m, 32);
      int   oi = __shfl_xor(bestI[r], m, 32);
      bool take = (ov < bestV[r]) || (ov == bestV[r] && oi < bestI[r]);
      bestV[r] = take ? ov : bestV[r];
      bestI[r] = take ? oi : bestI[r];
    }
  }
  if (lrow == 0) {
    #pragma unroll
    for (int r = 0; r < 8; ++r)
      enc[row0 + khalf * 8 + r] = bestI[r];   // lanes 0/16 -> rows 0-7 / 8-15
  }
}

// ---------------------------------------------------------------------------
// Kernel 3: gather fp32 codes, straight-through output, per-block sq-err sums
// ---------------------------------------------------------------------------
__global__ __launch_bounds__(256) void vq_gather(const float* __restrict__ x,
                                                 const float* __restrict__ ET,
                                                 const int* __restrict__ enc,
                                                 float* __restrict__ out,
                                                 float* __restrict__ partial) {
  __shared__ float wsum[8];
  const int t = threadIdx.x;
  const int row = blockIdx.x * 2 + (t >> 7);
  const int d = t & 127;
  const int idx = enc[row];
  const float q  = ET[(size_t)idx * DDIM + d];
  const float xv = x[(size_t)row * DDIM + d];
  out[(size_t)row * DDIM + d] = xv + (q - xv);     // match reference rounding
  const float diff = q - xv;
  float ss = diff * diff;
  #pragma unroll
  for (int m = 16; m >= 1; m >>= 1) ss += __shfl_xor(ss, m, 32);
  if ((t & 31) == 0) wsum[t >> 5] = ss;
  __syncthreads();
  if (t == 0) {
    float s = 0.f;
    #pragma unroll
    for (int w = 0; w < 8; ++w) s += wsum[w];
    partial[blockIdx.x] = s;
  }
}

// ---------------------------------------------------------------------------
// Kernel 4: deterministic final reduction -> loss = 1.25 * mean(sq)
// ---------------------------------------------------------------------------
__global__ __launch_bounds__(256) void vq_loss(const float* __restrict__ partial,
                                               float* __restrict__ loss) {
  __shared__ float wsum[8];
  const int t = threadIdx.x;
  float s = 0.f;
  for (int i = t; i < GATHER_BLOCKS; i += 256) s += partial[i];
  #pragma unroll
  for (int m = 16; m >= 1; m >>= 1) s += __shfl_xor(s, m, 32);
  if ((t & 31) == 0) wsum[t >> 5] = s;
  __syncthreads();
  if (t == 0) {
    float tot = 0.f;
    #pragma unroll
    for (int w = 0; w < 8; ++w) tot += wsum[w];
    *loss = 1.25f * tot / (float)(NROWS * DDIM);
  }
}

// ---------------------------------------------------------------------------
extern "C" void kernel_launch(void* const* d_in, const int* in_sizes, int n_in,
                              void* d_out, int out_size, void* d_ws, size_t ws_size,
                              hipStream_t stream) {
  const float* x = (const float*)d_in[0];       // [32,32,32,128] fp32
  const float* E = (const float*)d_in[1];       // [128,2048] fp32
  float* out  = (float*)d_out;                  // 4194304 floats + 1 loss
  float* loss = out + (size_t)NROWS * DDIM;

  float*  hn      = (float*)d_ws;                         // 2048 f
  float*  ET      = hn + KCODES;                          // 262144 f
  int*    enc     = (int*)(ET + (size_t)KCODES * DDIM);   // 32768 i32
  float*  partial = (float*)(enc + NROWS);                // 16384 f
  __bf16* Bsw     = (__bf16*)(partial + GATHER_BLOCKS);   // 524288 bf16 (1 MB)

  vq_prep  <<<KCODES / 256, 256, 0, stream>>>(E, hn, ET, Bsw);
  vq_argmin<<<NROWS / ROWS_PER_WG, AM_THREADS, 0, stream>>>(x, Bsw, hn, enc);
  vq_gather<<<GATHER_BLOCKS, 256, 0, stream>>>(x, ET, enc, out, partial);
  vq_loss  <<<1, 256, 0, stream>>>(partial, loss);
}
